// BaseAttention_63144609186016
// MI455X (gfx1250) — matrix operand order repaired and checked by reading, not compiled, not run on hardware
//
#include <hip/hip_runtime.h>
#include <hip/hip_bf16.h>

// ---------------------------------------------------------------------------
// MI455X (gfx1250) multi-head causal attention, f16 WMMA pipeline.
//   D_IN = D_OUT = 1024, N_HEADS = 16, HEAD_DIM = 64, B = 2, N = 2048
// Ping-pong pipelined WMMA GEMMs (async-to-LDS weight staging via
// GLOBAL_LOAD_ASYNC_TO_LDS_B128) + 32-row flash attention.
// ---------------------------------------------------------------------------

typedef __attribute__((ext_vector_type(16))) _Float16 v16h;
typedef __attribute__((ext_vector_type(8)))  _Float16 v8h;
typedef __attribute__((ext_vector_type(8)))  float    v8f;
typedef __attribute__((ext_vector_type(4)))  int      v4i;

#define AS1 __attribute__((address_space(1)))
#define AS3 __attribute__((address_space(3)))

#define D_MODEL 1024
#define SEQ     2048
#define BATCH   2
#define NHEADS  16
#define HDIM    64
#define MTOT    (BATCH * SEQ)   // 4096 rows

// B-tile LDS staging: 64 rows x 32 k-halfs, padded to 40 halfs (80 B) so the
// 16 lanes of a B-fragment read 16 distinct banks (16*80/4 mod 64 distinct).
#define BLD_PITCH 40

#if __has_builtin(__builtin_amdgcn_global_load_async_to_lds_b128)
#define USE_ASYNC_LDS 1
#else
#define USE_ASYNC_LDS 0
#endif

#if __has_builtin(__builtin_amdgcn_s_wait_asynccnt)
#define WAIT_ASYNC(n) __builtin_amdgcn_s_wait_asynccnt(n)
#else
#define WAIT_ASYNC(n) asm volatile("s_wait_asynccnt %0" ::"i"(n) : "memory")
#endif

// ----- WMMA wrapper: D = A(16x32 f16) * B(32x16 f16) + C(16x16 f32) --------
__device__ __forceinline__ v8f wmma_f16(v16h a, v16h b, v8f c) {
  return __builtin_amdgcn_wmma_f32_16x16x32_f16(
      /*neg_a=*/false, a, /*neg_b=*/false, b,
      /*c_mod=*/(short)0, c, /*reuse_a=*/false, /*reuse_b=*/false);
}

// ----- A fragment (16x32, MxK) from row-major f16, row stride lda ----------
// lane<16: row=lane,    K = 0..7  and 16..23
// lane>=16: row=lane-16, K = 8..15 and 24..31
__device__ __forceinline__ v16h load_a_frag(const _Float16* base, int lda) {
  const int lane = threadIdx.x & 31;
  const int row  = lane & 15;
  const int koff = (lane < 16) ? 0 : 8;
  const _Float16* p = base + row * lda + koff;
  v8h lo = *(const v8h*)(p);
  v8h hi = *(const v8h*)(p + 16);
  v16h f;
#pragma unroll
  for (int i = 0; i < 8; ++i) { f[i] = lo[i]; f[i + 8] = hi[i]; }
  return f;
}

// ----- B fragment (32x16, KxN) from BT stored row-major as (N x K) ---------
// lane<16: col=lane,    K = 0..7  and 16..23
// lane>=16: col=lane-16, K = 8..15 and 24..31
__device__ __forceinline__ v16h load_b_frag(const _Float16* baseT, int ldb) {
  const int lane = threadIdx.x & 31;
  const int col  = lane & 15;
  const int koff = (lane < 16) ? 0 : 8;
  const _Float16* p = baseT + col * ldb + koff;
  v8h lo = *(const v8h*)(p);
  v8h hi = *(const v8h*)(p + 16);
  v16h f;
#pragma unroll
  for (int i = 0; i < 8; ++i) { f[i] = lo[i]; f[i + 8] = hi[i]; }
  return f;
}

#if USE_ASYNC_LDS
// One async 16-byte-per-lane copy: global -> LDS (tracked by ASYNCcnt).
__device__ __forceinline__ void async_cp16(const _Float16* g, _Float16* l) {
  void* gv = (void*)g;   // drop const
  void* lv = (void*)l;
  __builtin_amdgcn_global_load_async_to_lds_b128(
      (AS1 v4i*)gv, (AS3 v4i*)lv, 0, 0);
}

// Stage a 64-row x 32-half B tile (BT rows n0..n0+63, k = kcol..kcol+31) into
// LDS via async DMA: 8 x GLOBAL_LOAD_ASYNC_TO_LDS_B128 (each lane 16 bytes).
__device__ __forceinline__ void stage_bt(const _Float16* BT, int n0, int kcol,
                                         _Float16* dst) {
  const int lane = threadIdx.x & 31;
#pragma unroll
  for (int i = 0; i < 8; ++i) {
    int idx = i * 32 + lane;
    int row = idx >> 2;   // 0..63
    int c   = idx & 3;    // 16-byte chunk within the 64-byte row
    async_cp16(BT + (size_t)(n0 + row) * D_MODEL + kcol + 8 * c,
               dst + row * BLD_PITCH + 8 * c);
  }
}
#endif

// One K-step of the 32x64 tile: 4 B fragments + 8 WMMAs.
__device__ __forceinline__ void gemm_phase(const v16h a[2],
                                           const _Float16* Bbase, int ldb,
                                           v8f acc[2][4]) {
  v16h b[4];
#pragma unroll
  for (int j = 0; j < 4; ++j)
    b[j] = load_b_frag(Bbase + (size_t)(16 * j) * ldb, ldb);
#pragma unroll
  for (int i = 0; i < 2; ++i)
#pragma unroll
    for (int j = 0; j < 4; ++j)
      acc[i][j] = wmma_f16(a[i], b[j], acc[i][j]);
}

// ---------------------------------------------------------------------------
// Ping-pong pipelined 32x64 register-tile GEMM: acc[2][4] (rows x cols).
// A row-major (lda = D_MODEL), BT row-major (N x K), K = D_MODEL.
__device__ __forceinline__ void gemm_32x64(const _Float16* __restrict__ A,
                                           const _Float16* __restrict__ BT,
                                           int m0, int n0, v8f acc[2][4]) {
  const _Float16* A0 = A + (size_t)m0 * D_MODEL;
  const _Float16* A1 = A + (size_t)(m0 + 16) * D_MODEL;
#if USE_ASYNC_LDS
  __shared__ _Float16 Bld[2][64 * BLD_PITCH];
  v16h a0[2], a1[2];
  stage_bt(BT, n0, 0, &Bld[0][0]);          // buf0 <- k 0..31
  a0[0] = load_a_frag(A0, D_MODEL);
  a0[1] = load_a_frag(A1, D_MODEL);
#pragma unroll 1
  for (int kk = 0; kk < D_MODEL; kk += 64) {
    // phase 0: compute k=kk from buf0/a0; prefetch k=kk+32 -> buf1/a1
    stage_bt(BT, n0, kk + 32, &Bld[1][0]);  // kk+32 <= 992, always valid
    a1[0] = load_a_frag(A0 + kk + 32, D_MODEL);
    a1[1] = load_a_frag(A1 + kk + 32, D_MODEL);
    WAIT_ASYNC(8);                           // buf0 ready, buf1 in flight
    gemm_phase(a0, &Bld[0][0], BLD_PITCH, acc);
    // phase 1: compute k=kk+32 from buf1/a1; prefetch k=kk+64 -> buf0/a0
    if (kk + 64 < D_MODEL) {
      stage_bt(BT, n0, kk + 64, &Bld[0][0]);
      a0[0] = load_a_frag(A0 + kk + 64, D_MODEL);
      a0[1] = load_a_frag(A1 + kk + 64, D_MODEL);
      WAIT_ASYNC(8);
    } else {
      WAIT_ASYNC(0);
    }
    gemm_phase(a1, &Bld[1][0], BLD_PITCH, acc);
  }
#else
  // Register-only ping-pong (no rotation copies: loads write alternate set).
  v16h a0[2], a1[2], b0[4], b1[4];
  a0[0] = load_a_frag(A0, D_MODEL);
  a0[1] = load_a_frag(A1, D_MODEL);
#pragma unroll
  for (int j = 0; j < 4; ++j)
    b0[j] = load_b_frag(BT + (size_t)(n0 + 16 * j) * D_MODEL, D_MODEL);
#pragma unroll 1
  for (int kk = 0; kk < D_MODEL; kk += 64) {
    // phase 0: compute kk from set0; load kk+32 into set1
    a1[0] = load_a_frag(A0 + kk + 32, D_MODEL);
    a1[1] = load_a_frag(A1 + kk + 32, D_MODEL);
#pragma unroll
    for (int j = 0; j < 4; ++j)
      b1[j] = load_b_frag(BT + (size_t)(n0 + 16 * j) * D_MODEL + kk + 32,
                          D_MODEL);
#pragma unroll
    for (int i = 0; i < 2; ++i)
#pragma unroll
      for (int j = 0; j < 4; ++j)
        acc[i][j] = wmma_f16(a0[i], b0[j], acc[i][j]);
    // phase 1: compute kk+32 from set1; load kk+64 into set0 (if any)
    if (kk + 64 < D_MODEL) {
      a0[0] = load_a_frag(A0 + kk + 64, D_MODEL);
      a0[1] = load_a_frag(A1 + kk + 64, D_MODEL);
#pragma unroll
      for (int j = 0; j < 4; ++j)
        b0[j] = load_b_frag(BT + (size_t)(n0 + 16 * j) * D_MODEL + kk + 64,
                            D_MODEL);
    }
#pragma unroll
    for (int i = 0; i < 2; ++i)
#pragma unroll
      for (int j = 0; j < 4; ++j)
        acc[i][j] = wmma_f16(a1[i], b1[j], acc[i][j]);
  }
#endif
}

// ---------------------------------------------------------------------------
// Prep: fp32 x -> f16
__global__ __launch_bounds__(256) void convert_x_kernel(
    const float* __restrict__ x, _Float16* __restrict__ xh) {
  int i = blockIdx.x * 256 + threadIdx.x;
  xh[i] = (_Float16)x[i];
}

// Prep: fp32 W (K x N) -> f16 transposed WT (N x K); blockIdx.y picks weight.
__global__ __launch_bounds__(256) void prep_weights_kernel(
    const float* __restrict__ Wq, const float* __restrict__ Wk,
    const float* __restrict__ Wv, const float* __restrict__ Wo,
    _Float16* __restrict__ WqT, _Float16* __restrict__ WkT,
    _Float16* __restrict__ WvT, _Float16* __restrict__ WoT) {
  int idx = blockIdx.x * 256 + threadIdx.x;        // 0 .. 1M-1
  int k = idx >> 10, n = idx & 1023;
  const float* src; _Float16* dst;
  switch (blockIdx.y) {
    case 0:  src = Wq; dst = WqT; break;
    case 1:  src = Wk; dst = WkT; break;
    case 2:  src = Wv; dst = WvT; break;
    default: src = Wo; dst = WoT; break;
  }
  dst[n * D_MODEL + k] = (_Float16)src[k * D_MODEL + n];
}

// ---------------------------------------------------------------------------
// QKV projection: one wave computes a 32x64 tile of x @ W.
// which = blockIdx.z: 0 -> Qh (row-major), 1 -> Kh (row-major),
//                     2 -> Vt (transposed: [b][feature][token])
__global__ __launch_bounds__(32) void qkv_gemm_kernel(
    const _Float16* __restrict__ xh,
    const _Float16* __restrict__ WqT, const _Float16* __restrict__ WkT,
    const _Float16* __restrict__ WvT,
    _Float16* __restrict__ Qh, _Float16* __restrict__ Kh,
    _Float16* __restrict__ Vt) {
  const int m0    = blockIdx.x * 32;
  const int n0    = blockIdx.y * 64;
  const int which = blockIdx.z;
  const _Float16* WT = (which == 0) ? WqT : (which == 1) ? WkT : WvT;
  const int lane = threadIdx.x & 31;

  v8f acc[2][4] = {};
  gemm_32x64(xh, WT, m0, n0, acc);

  const int r8 = (lane < 16) ? 0 : 8;
  const int cn = lane & 15;
#pragma unroll
  for (int i = 0; i < 2; ++i)
#pragma unroll
    for (int j = 0; j < 4; ++j)
#pragma unroll
      for (int r = 0; r < 8; ++r) {
        int row = m0 + 16 * i + r8 + r;
        int n   = n0 + 16 * j + cn;
        _Float16 v = (_Float16)acc[i][j][r];
        if (which == 0) {
          Qh[(size_t)row * D_MODEL + n] = v;
        } else if (which == 1) {
          Kh[(size_t)row * D_MODEL + n] = v;
        } else {
          int b = row >> 11, t = row & (SEQ - 1);
          Vt[(size_t)b * D_MODEL * SEQ + (size_t)n * SEQ + t] = v;
        }
      }
}

// ---------------------------------------------------------------------------
// Flash attention: one wave per (batch*head, 32-row q tile).
// Streams keys in tiles of 32; K/V fragments are shared by two 16-row
// S-tiles (2x arithmetic intensity); online softmax in exp2 domain.
__global__ __launch_bounds__(32) void flash_attn_kernel(
    const _Float16* __restrict__ Qh, const _Float16* __restrict__ Kh,
    const _Float16* __restrict__ Vt, _Float16* __restrict__ Ctx) {
  __shared__ _Float16 Plds[2][16 * 32];

  const int qt    = blockIdx.x;            // 0..63
  const int bh    = blockIdx.y;            // 0..31
  const int bb    = bh >> 4, h = bh & 15;
  const int qbase = qt * 32;
  const int lane  = threadIdx.x & 31;
  const int r8    = (lane < 16) ? 0 : 8;
  const int cn    = lane & 15;

  v16h a[2][2];
#pragma unroll
  for (int u = 0; u < 2; ++u) {
    const _Float16* qptr =
        Qh + ((size_t)(bb * SEQ + qbase + 16 * u)) * D_MODEL + h * HDIM;
    a[u][0] = load_a_frag(qptr,      D_MODEL);   // K = d 0..31
    a[u][1] = load_a_frag(qptr + 32, D_MODEL);   // K = d 32..63
  }

  v8f acc[2][4] = {};
  float mi[2][8], li[2][8];
#pragma unroll
  for (int u = 0; u < 2; ++u)
#pragma unroll
    for (int r = 0; r < 8; ++r) { mi[u][r] = -1e30f; li[u][r] = 0.0f; }

  // scores scaled by 1/sqrt(64)=1/8 and converted to log2 domain
  const float sc = 0.125f * 1.44269504088896340736f;

#pragma unroll 1
  for (int kb = 0; kb <= qbase; kb += 32) {
    // ---- issue ALL of this tile's operand loads up front ----
    v16h bv[4];                                       // V: consumed last
#pragma unroll
    for (int j = 0; j < 4; ++j)
      bv[j] = load_b_frag(
          Vt + ((size_t)(bb * D_MODEL + h * HDIM + 16 * j)) * SEQ + kb, SEQ);
    v16h bk[2][2];                                    // K^T fragments
#pragma unroll
    for (int t = 0; t < 2; ++t) {
      const _Float16* kp =
          Kh + ((size_t)(bb * SEQ + kb + 16 * t)) * D_MODEL + h * HDIM;
      bk[t][0] = load_b_frag(kp,      D_MODEL);
      bk[t][1] = load_b_frag(kp + 32, D_MODEL);
    }
    // ---- prefetch next kv tile (global_prefetch_b8) ----
    if (kb + 32 <= qbase) {
      __builtin_prefetch(
          Kh + (size_t)(bb * SEQ + kb + 32 + lane) * D_MODEL + h * HDIM, 0, 3);
      __builtin_prefetch(
          Vt + ((size_t)(bb * D_MODEL + h * HDIM + 2 * lane)) * SEQ + kb + 32,
          0, 3);
    }

    // ---- S = Q K^T : 8 WMMAs (2 row-tiles x 2 col-tiles x K-dim 64) ----
    v8f s[2][2];
#pragma unroll
    for (int u = 0; u < 2; ++u)
#pragma unroll
      for (int t = 0; t < 2; ++t) {
        v8f z = {};
        z       = wmma_f16(a[u][0], bk[t][0], z);
        s[u][t] = wmma_f16(a[u][1], bk[t][1], z);
      }

    // ---- scale + causal mask + online softmax (per row-tile u) ----
#pragma unroll
    for (int u = 0; u < 2; ++u) {
      float tm[8];
#pragma unroll
      for (int r = 0; r < 8; ++r) {
        int rowg = qbase + 16 * u + r8 + r;
        float v0 = s[u][0][r] * sc; if (kb + cn      > rowg) v0 = -1e30f;
        float v1 = s[u][1][r] * sc; if (kb + 16 + cn > rowg) v1 = -1e30f;
        s[u][0][r] = v0; s[u][1][r] = v1;
        tm[r] = fmaxf(v0, v1);
      }
#pragma unroll
      for (int m = 1; m <= 8; m <<= 1)
#pragma unroll
        for (int r = 0; r < 8; ++r)
          tm[r] = fmaxf(tm[r], __shfl_xor(tm[r], m, 32));
      float rs[8];
#pragma unroll
      for (int r = 0; r < 8; ++r) {
        float mnew = fmaxf(mi[u][r], tm[r]);
        float fac  = exp2f(mi[u][r] - mnew);
        float p0   = exp2f(s[u][0][r] - mnew);
        float p1   = exp2f(s[u][1][r] - mnew);
        s[u][0][r] = p0; s[u][1][r] = p1;
        rs[r]      = p0 + p1;
        li[u][r]  *= fac;
        mi[u][r]   = mnew;
#pragma unroll
        for (int j = 0; j < 4; ++j) acc[u][j][r] *= fac;
      }
#pragma unroll
      for (int m = 1; m <= 8; m <<= 1)
#pragma unroll
        for (int r = 0; r < 8; ++r)
          rs[r] += __shfl_xor(rs[r], m, 32);
#pragma unroll
      for (int r = 0; r < 8; ++r) li[u][r] += rs[r];

      // ---- re-layout P (C layout -> A fragment) through LDS ----
#pragma unroll
      for (int t = 0; t < 2; ++t)
#pragma unroll
        for (int r = 0; r < 8; ++r)
          Plds[u][(r8 + r) * 32 + 16 * t + cn] = (_Float16)s[u][t][r];
    }
    asm volatile("s_wait_dscnt 0" ::: "memory");
    v16h ap[2];
#pragma unroll
    for (int u = 0; u < 2; ++u) ap[u] = load_a_frag(&Plds[u][0], 32);
    asm volatile("" ::: "memory");

    // ---- ctx += P @ V : 8 WMMAs ----
#pragma unroll
    for (int u = 0; u < 2; ++u)
#pragma unroll
      for (int j = 0; j < 4; ++j)
        acc[u][j] = wmma_f16(ap[u], bv[j], acc[u][j]);
    asm volatile("s_wait_dscnt 0" ::: "memory");
  }

  // ---- normalize and store ctx (f16, row-major (b, tok, 1024)) ----
#pragma unroll
  for (int u = 0; u < 2; ++u) {
#pragma unroll
    for (int r = 0; r < 8; ++r) li[u][r] = 1.0f / li[u][r];
#pragma unroll
    for (int j = 0; j < 4; ++j)
#pragma unroll
      for (int r = 0; r < 8; ++r) {
        size_t row = (size_t)(bb * SEQ + qbase + 16 * u + r8 + r);
        Ctx[row * D_MODEL + h * HDIM + 16 * j + cn] =
            (_Float16)(acc[u][j][r] * li[u][r]);
      }
  }
}

// ---------------------------------------------------------------------------
// Output projection: out = ctx @ W_o + b_o (f32 out), 32x64 tile per wave.
__global__ __launch_bounds__(32) void out_gemm_kernel(
    const _Float16* __restrict__ Ctx, const _Float16* __restrict__ WoT,
    const float* __restrict__ bo, float* __restrict__ out) {
  const int m0 = blockIdx.x * 32;
  const int n0 = blockIdx.y * 64;
  const int lane = threadIdx.x & 31;

  v8f acc[2][4] = {};
  gemm_32x64(Ctx, WoT, m0, n0, acc);

  const int r8 = (lane < 16) ? 0 : 8;
  const int cn = lane & 15;
#pragma unroll
  for (int i = 0; i < 2; ++i)
#pragma unroll
    for (int j = 0; j < 4; ++j)
#pragma unroll
      for (int r = 0; r < 8; ++r) {
        int row = m0 + 16 * i + r8 + r;
        int n   = n0 + 16 * j + cn;
        out[(size_t)row * D_MODEL + n] = acc[i][j][r] + bo[n];
      }
}

// ---------------------------------------------------------------------------
extern "C" void kernel_launch(void* const* d_in, const int* in_sizes, int n_in,
                              void* d_out, int out_size, void* d_ws,
                              size_t ws_size, hipStream_t stream) {
  const float* x  = (const float*)d_in[0];
  const float* Wq = (const float*)d_in[1];
  const float* Wk = (const float*)d_in[2];
  const float* Wv = (const float*)d_in[3];
  const float* Wo = (const float*)d_in[4];
  const float* bo = (const float*)d_in[5];
  float* out = (float*)d_out;

  char* ws = (char*)d_ws;
  const size_t MB = (size_t)1 << 20;
  _Float16* xh  = (_Float16*)(ws +  0 * MB);   // 8 MB  (4096 x 1024)
  _Float16* WqT = (_Float16*)(ws +  8 * MB);   // 2 MB
  _Float16* WkT = (_Float16*)(ws + 10 * MB);   // 2 MB
  _Float16* WvT = (_Float16*)(ws + 12 * MB);   // 2 MB
  _Float16* WoT = (_Float16*)(ws + 14 * MB);   // 2 MB
  _Float16* Qh  = (_Float16*)(ws + 16 * MB);   // 8 MB  (b, tok, 1024)
  _Float16* Kh  = (_Float16*)(ws + 24 * MB);   // 8 MB  (b, tok, 1024)
  _Float16* Vt  = (_Float16*)(ws + 32 * MB);   // 8 MB  (b, 1024, tok)
  _Float16* Ctx = (_Float16*)(ws + 40 * MB);   // 8 MB  (b, tok, 1024)

  // 1) precision / layout prep
  convert_x_kernel<<<(MTOT * D_MODEL) / 256, 256, 0, stream>>>(x, xh);
  prep_weights_kernel<<<dim3((D_MODEL * D_MODEL) / 256, 4), 256, 0, stream>>>(
      Wq, Wk, Wv, Wo, WqT, WkT, WvT, WoT);

  // 2) fused QKV projections (WMMA, 32x64 tiles, async-LDS pipelined)
  qkv_gemm_kernel<<<dim3(MTOT / 32, D_MODEL / 64, 3), 32, 0, stream>>>(
      xh, WqT, WkT, WvT, Qh, Kh, Vt);

  // 3) causal flash attention (WMMA + online softmax, 32-row q tiles)
  flash_attn_kernel<<<dim3(SEQ / 32, BATCH * NHEADS), 32, 0, stream>>>(
      Qh, Kh, Vt, Ctx);

  // 4) output projection + bias (WMMA, 32x64 tiles, async-LDS pipelined)
  out_gemm_kernel<<<dim3(MTOT / 32, D_MODEL / 64), 32, 0, stream>>>(
      Ctx, WoT, bo, out);
}
